// LinearRNN_24996709663059
// MI455X (gfx1250) — matrix operand order repaired and checked by reading, not compiled, not run on hardware
//
#include <hip/hip_runtime.h>

#define B_ 64
#define T_ 512
#define D_ 512
#define H_ 1024
#define KC 256   // K-chunk staged in LDS per stage phase
#define NCHUNK (H_ / KC)

typedef __attribute__((ext_vector_type(16))) __bf16 v16bf;
typedef __attribute__((ext_vector_type(8)))  float  v8f;
typedef __attribute__((ext_vector_type(4)))  unsigned int v4u;
typedef __attribute__((ext_vector_type(8)))  unsigned int v8u;

typedef __attribute__((address_space(3))) unsigned short lds_ushort;

// Native f32 -> bf16 conversion (hardware cvt on gfx1250)
static __device__ __forceinline__ unsigned short f2bf16(float f) {
    return __builtin_bit_cast(unsigned short, (__bf16)f);
}

// One packed v_cvt_pk_bf16_f32 when available
#if __has_builtin(__builtin_amdgcn_cvt_pk_bf16_f32)
static __device__ __forceinline__ unsigned int pack_bf16(float a, float b) {
    auto p = __builtin_amdgcn_cvt_pk_bf16_f32(a, b);
    return __builtin_bit_cast(unsigned int, p);
}
#else
static __device__ __forceinline__ unsigned int pack_bf16(float a, float b) {
    return (unsigned)f2bf16(a) | ((unsigned)f2bf16(b) << 16);
}
#endif

static __device__ __forceinline__ v16bf mkfrag(v4u lo, v4u hi) {
    v8u t;
    t[0] = lo[0]; t[1] = lo[1]; t[2] = lo[2]; t[3] = lo[3];
    t[4] = hi[0]; t[5] = hi[1]; t[6] = hi[2]; t[7] = hi[3];
    return __builtin_bit_cast(v16bf, t);
}

// Async DMA: 16 bytes global -> LDS, tracked by ASYNCcnt (CDNA5)
static __device__ __forceinline__ void async_copy_b128(unsigned short* lds_dst,
                                                       const unsigned short* gsrc) {
    lds_ushort* lp = (lds_ushort*)lds_dst;
    asm volatile("global_load_async_to_lds_b128 %0, %1, off"
                 :: "v"(lp), "v"(gsrc)
                 : "memory");
}

static __device__ __forceinline__ void wait_asynccnt0() {
    asm volatile("s_wait_asynccnt 0x0" ::: "memory");
}

// Convert weights + h0 to bf16 into workspace; zero the global barrier counter.
__global__ void prep_kernel(const float* __restrict__ Whh, const float* __restrict__ Wxh,
                            const float* __restrict__ h0,
                            unsigned short* __restrict__ whh_bf,
                            unsigned short* __restrict__ wxh_bf,
                            unsigned short* __restrict__ hbf0,
                            unsigned int* __restrict__ counter) {
    int idx = blockIdx.x * blockDim.x + threadIdx.x;
    if (idx == 0) *counter = 0u;
    const int nhh = H_ * H_;
    const int nxh = H_ * D_;
    if (idx < nhh) {
        whh_bf[idx] = f2bf16(Whh[idx]);
    } else if (idx < nhh + nxh) {
        wxh_bf[idx - nhh] = f2bf16(Wxh[idx - nhh]);
    } else if (idx < nhh + nxh + B_ * H_) {
        int j = idx - nhh - nxh;
        hbf0[j] = f2bf16(h0[j]);
    }
}

// Phase 1: out[m, n] = sum_k x[m,k] * Wxh[n,k] + b[n], M = B*T rows flat.
// Block = 512 threads = 16 waves; 64(M) x 64(N) tile; each wave one 16x16 C tile.
// Stage 64xKC f32->bf16 slab, then KC/32 WMMAs per barrier pair.
__global__ __launch_bounds__(512) void xw_gemm_kernel(
    const float* __restrict__ x, const float* __restrict__ bias,
    const unsigned short* __restrict__ wxh_bf, float* __restrict__ out)
{
    __shared__ alignas(16) unsigned short sA[64 * KC];   // 32 KB
    const int tid  = threadIdx.x;
    const int wave = tid >> 5, lane = tid & 31, l15 = lane & 15, hi = lane >> 4;
    const int wm = wave >> 2, wn = wave & 3;
    const int m_base = blockIdx.x * 64;
    const int nlane  = blockIdx.y * 64 + wn * 16 + l15;

    v8f c;
    {
        float bv = bias[nlane];
        #pragma unroll
        for (int r = 0; r < 8; ++r) c[r] = bv;
    }

    const int sm = tid >> 3;          // staging row 0..63
    const int sk = (tid & 7) * 32;    // staging col base (32 elems per thread)

    for (int kk0 = 0; kk0 < D_; kk0 += KC) {
        const float* xp = x + (size_t)(m_base + sm) * D_ + kk0 + sk;
        unsigned int* sp = (unsigned int*)&sA[sm * KC + sk];
        #pragma unroll
        for (int q = 0; q < 8; ++q) {
            float4 f = *(const float4*)(xp + q * 4);
            sp[2 * q]     = pack_bf16(f.x, f.y);
            sp[2 * q + 1] = pack_bf16(f.z, f.w);
        }
        __syncthreads();

        #pragma unroll
        for (int kk = 0; kk < KC; kk += 32) {
            const unsigned short* arow =
                &sA[(wm * 16 + l15) * KC + kk + (hi ? 8 : 0)];
            v16bf av = mkfrag(*(const v4u*)arow, *(const v4u*)(arow + 16));
            const unsigned short* brow =
                wxh_bf + (size_t)nlane * D_ + kk0 + kk + (hi ? 8 : 0);
            v16bf bv = mkfrag(*(const v4u*)brow, *(const v4u*)(brow + 16));
            c = __builtin_amdgcn_wmma_f32_16x16x32_bf16(false, av, false, bv,
                                                        (short)0, c, false, false);
        }
        __syncthreads();
    }

    #pragma unroll
    for (int r = 0; r < 8; ++r) {
        int m = m_base + wm * 16 + hi * 8 + r;
        out[(size_t)m * H_ + nlane] = c[r];
    }
}

// Phase 2: persistent scan. 16 workgroups, each owns a 64-column slice of H.
// The block's 64x1024 bf16 W_hh slice is loaded into LDS ONCE (async DMA) and
// reused for all 512 steps. h is double-buffered in bf16 in ws; the staging
// into LDS is ping-ponged so the async DMA for chunk k+1 overlaps the WMMAs
// of chunk k. Inner loop: 4x ds_load_b128 + 1x v_wmma, no VMEM.
__global__ __launch_bounds__(512) void scan_kernel(
    const unsigned short* __restrict__ whh_bf,
    unsigned short* __restrict__ hbf0, unsigned short* __restrict__ hbf1,
    float* __restrict__ out, float* __restrict__ hlast,
    unsigned int* __restrict__ counter)
{
    __shared__ alignas(16) unsigned short sW[64 * H_];      // 128 KB: W slice
    __shared__ alignas(16) unsigned short sA[2][64 * KC];   // 2 x 32 KB: h staging
    const int tid  = threadIdx.x;
    const int wave = tid >> 5, lane = tid & 31, l15 = lane & 15, hi = lane >> 4;
    const int wm = wave >> 2, wn = wave & 3;
    const int n0 = blockIdx.x * 64;
    const int nl = wn * 16 + l15;          // local column in W slice
    const int nlane = n0 + nl;             // global column
    const size_t rowstride = (size_t)T_ * H_;  // batch stride in out

    // Preload W slice into LDS: 64x1024 bf16 = 8192 b128 chunks, 16 per thread
    #pragma unroll
    for (int j = 0; j < 16; ++j) {
        int cidx = tid + 512 * j;          // 0..8191
        int r = cidx >> 7;                 // 128 chunks per row
        int k = (cidx & 127) * 8;
        async_copy_b128(&sW[r * H_ + k], whh_bf + (size_t)(n0 + r) * H_ + k);
    }
    wait_asynccnt0();
    __syncthreads();

    // stage 64 x KC bf16 (32 KB) from hr[:, kc0:kc0+KC] into sA[buf]
    auto stage = [&](const unsigned short* hr, int kc0, int buf) {
        #pragma unroll
        for (int j = 0; j < 4; ++j) {
            int cidx = tid + 512 * j;          // 0..2047
            int m    = cidx >> 5;              // 32 chunks per row
            int kc   = (cidx & 31) * 8;
            async_copy_b128(&sA[buf][m * KC + kc],
                            hr + (size_t)m * H_ + kc0 + kc);
        }
    };

    for (int t = 0; t < T_; ++t) {
        const unsigned short* hr = (t & 1) ? hbf1 : hbf0;   // read h_{t-1} (bf16)
        unsigned short*       hw = (t & 1) ? hbf0 : hbf1;   // write h_t (bf16)
        float* cur = out + (size_t)t * H_;

        // init accumulator with xw_t
        v8f c;
        #pragma unroll
        for (int r = 0; r < 8; ++r) {
            int m = wm * 16 + hi * 8 + r;
            c[r] = cur[(size_t)m * rowstride + nlane];
        }

        stage(hr, 0, 0);
        wait_asynccnt0();
        __syncthreads();

        #pragma unroll
        for (int chunk = 0; chunk < NCHUNK; ++chunk) {
            const int kc0 = chunk * KC;
            // prefetch next chunk while computing this one
            if (chunk + 1 < NCHUNK) stage(hr, kc0 + KC, (chunk + 1) & 1);

            const unsigned short* sAc = &sA[chunk & 1][0];
            #pragma unroll
            for (int kk = 0; kk < KC; kk += 32) {
                const unsigned short* arow =
                    sAc + (wm * 16 + l15) * KC + kk + (hi ? 8 : 0);
                v16bf av = mkfrag(*(const v4u*)arow, *(const v4u*)(arow + 16));
                const unsigned short* brow =
                    &sW[nl * H_ + kc0 + kk + (hi ? 8 : 0)];
                v16bf bv = mkfrag(*(const v4u*)brow, *(const v4u*)(brow + 16));
                c = __builtin_amdgcn_wmma_f32_16x16x32_bf16(false, av, false, bv,
                                                            (short)0, c, false, false);
            }
            wait_asynccnt0();   // next chunk's DMA has landed
            __syncthreads();
        }

        // store h_t: f32 into out, bf16 into next-step buffer
        #pragma unroll
        for (int r = 0; r < 8; ++r) {
            int m = wm * 16 + hi * 8 + r;
            cur[(size_t)m * rowstride + nlane] = c[r];
            hw[(size_t)m * H_ + nlane] = f2bf16(c[r]);
            if (t == T_ - 1) hlast[(size_t)m * H_ + nlane] = c[r];
        }

        // device-wide barrier between steps (16 resident workgroups)
        if (t < T_ - 1) {
            __threadfence();
            __syncthreads();
            if (tid == 0) {
                atomicAdd(counter, 1u);
                unsigned int target = 16u * (unsigned int)(t + 1);
                while (atomicAdd(counter, 0u) < target) __builtin_amdgcn_s_sleep(2);
            }
            __syncthreads();
            __threadfence();
        }
    }
}

extern "C" void kernel_launch(void* const* d_in, const int* in_sizes, int n_in,
                              void* d_out, int out_size, void* d_ws, size_t ws_size,
                              hipStream_t stream) {
    const float* x   = (const float*)d_in[0];   // [B,T,D]
    const float* h0  = (const float*)d_in[1];   // [B,H]
    const float* Whh = (const float*)d_in[2];   // [H,H]
    const float* Wxh = (const float*)d_in[3];   // [H,D]
    const float* b_h = (const float*)d_in[4];   // [H]

    float* out   = (float*)d_out;                       // [B,T,H]
    float* hlast = out + (size_t)B_ * T_ * H_;          // [B,H]

    unsigned short* whh_bf = (unsigned short*)d_ws;             // 2 MB @ 0
    unsigned short* wxh_bf = whh_bf + (size_t)H_ * H_;          // 1 MB @ 2 MB
    unsigned short* hbf0   = wxh_bf + (size_t)H_ * D_;          // 128 KB @ 3 MB
    unsigned short* hbf1   = hbf0 + (size_t)B_ * H_;            // 128 KB
    unsigned int*   counter = (unsigned int*)((char*)d_ws + (3u << 20) + (512u << 10));

    (void)in_sizes; (void)n_in; (void)out_size; (void)ws_size;

    const int nconv = H_ * H_ + H_ * D_ + B_ * H_;
    prep_kernel<<<(nconv + 255) / 256, 256, 0, stream>>>(Whh, Wxh, h0,
                                                         whh_bf, wxh_bf, hbf0, counter);
    xw_gemm_kernel<<<dim3((B_ * T_) / 64, H_ / 64), 512, 0, stream>>>(x, b_h, wxh_bf, out);
    scan_kernel<<<H_ / 64, 512, 0, stream>>>(whh_bf, hbf0, hbf1, out, hlast, counter);
}
